// CausalAttention_90228672954503
// MI455X (gfx1250) — matrix operand order repaired and checked by reading, not compiled, not run on hardware
//
#include <hip/hip_runtime.h>
#include <cmath>

// ---------------------------------------------------------------------------
// CausalAttention for MI455X (gfx1250, wave32, WMMA)
//   B=4, S=4096, D_IN=D_OUT=1024
//   K0a: x fp32 -> f16 (coalesced, once)
//   K0b: W fp32 [o][d] -> f16 W^T [d][o] (LDS tile transpose, once)
//   K1 : QKV projection GEMM, 128x128 tile, b128 A / strided-gather B LDS fragments,
//        async global->LDS staging (ASYNCcnt)
//   K2 : causal flash attention, S^T = K*Q^T, ~185KB LDS per workgroup,
//        async K/V tile streaming, b128 A / strided-gather B LDS fragments
// ---------------------------------------------------------------------------

typedef __attribute__((ext_vector_type(16))) _Float16 v16h;
typedef __attribute__((ext_vector_type(8)))  float    v8f;
typedef int v4i __attribute__((__vector_size__(4 * sizeof(int))));

#define BATCH 4
#define SEQ   4096
#define DIM   1024

// Probe for the CDNA5 async global->LDS copy builtin (ASYNCcnt-tracked).
#if defined(__has_builtin)
#if __has_builtin(__builtin_amdgcn_global_load_async_to_lds_b128)
#define USE_ASYNC_COPY 1
#endif
#endif
#ifndef USE_ASYNC_COPY
#define USE_ASYNC_COPY 0
#endif

__device__ __forceinline__ void async_b128(const _Float16* gsrc, _Float16* ldst) {
#if USE_ASYNC_COPY
  __builtin_amdgcn_global_load_async_to_lds_b128((v4i*)gsrc, (v4i*)ldst, 0, 0);
#else
  *(uint4*)ldst = *(const uint4*)gsrc;
#endif
}

__device__ __forceinline__ void async_wait() {
#if USE_ASYNC_COPY
  asm volatile("s_wait_asynccnt 0x0" ::: "memory");
#endif
}

// ---- WMMA helpers ---------------------------------------------------------

__device__ __forceinline__ v8f wmma_f16(v16h a, v16h b, v8f c) {
  return __builtin_amdgcn_wmma_f32_16x16x32_f16(
      /*neg_a=*/false, a, /*neg_b=*/false, b,
      /*c_mod=*/(short)0, c, /*reuse_a=*/false, /*reuse_b=*/false);
}

// A-matrix fragment, 16x32 f16 (M x K): lane l holds row m = l&15;
// halves 0..7 = K (l>>4)*8 .. +7, halves 8..15 = K 16+(l>>4)*8 .. +7.
// Two 16-byte chunks -> two ds_load_b128.
__device__ __forceinline__ v16h frag_a16(const _Float16* p, int ld, int lane) {
  union { v16h v; uint4 u[2]; } r;
  const _Float16* row = p + (lane & 15) * ld + (lane >> 4) * 8;
  r.u[0] = *(const uint4*)(row);
  r.u[1] = *(const uint4*)(row + 16);
  return r.v;
}

// B-matrix fragment, 32x16 f16 (K x N); lane holds column n = l&15;
// halves 0..7 = K (l>>4)*8 .. +7, halves 8..15 = K 16+(l>>4)*8 .. +7 (strided gather).
__device__ __forceinline__ v16h frag_b_nf(const _Float16* p, int ld, int lane) {
  v16h r;
  const _Float16* col = p + (lane & 15) + (lane >> 4) * 8 * ld;
#pragma unroll
  for (int s = 0; s < 8; ++s) {
    r[s] = col[s * ld];
    r[s + 8] = col[(16 + s) * ld];
  }
  return r;
}

// ---------------------------------------------------------------------------
// Kernel 0a: convert x (fp32) -> xh (f16), fully coalesced.
// ---------------------------------------------------------------------------

__global__ __launch_bounds__(256) void convert_x_kernel(
    const float* __restrict__ x, _Float16* __restrict__ xh) {
  const size_t i = ((size_t)blockIdx.x * 256 + threadIdx.x) * 4;
  const float4 v = *(const float4*)&x[i];
  union { _Float16 h[4]; uint2 u; } r;
  r.h[0] = (_Float16)v.x;
  r.h[1] = (_Float16)v.y;
  r.h[2] = (_Float16)v.z;
  r.h[3] = (_Float16)v.w;
  *(uint2*)&xh[i] = r.u;
}

// ---------------------------------------------------------------------------
// Kernel 0b: Wt[z][d][o] = (f16) W_z[o][d]  via 32x32 LDS tile transpose.
//   grid = (DIM/32, DIM/32, 3), block = 256.
// ---------------------------------------------------------------------------

__global__ __launch_bounds__(256) void transpose_w_kernel(
    const float* __restrict__ Wq, const float* __restrict__ Wk,
    const float* __restrict__ Wv, _Float16* __restrict__ Wt) {
  __shared__ _Float16 t[32 * 33];
  const float* W = (blockIdx.z == 0) ? Wq : (blockIdx.z == 1) ? Wk : Wv;
  _Float16* O = Wt + (size_t)blockIdx.z * DIM * DIM;
  const int d0 = blockIdx.x * 32;
  const int o0 = blockIdx.y * 32;
  const int tid = threadIdx.x;
#pragma unroll
  for (int it = 0; it < 4; ++it) {
    const int idx = tid + it * 256;  // 0..1023
    const int r = idx >> 5, c = idx & 31;
    t[r * 33 + c] = (_Float16)W[(size_t)(o0 + r) * DIM + d0 + c];  // read along d
  }
  __syncthreads();
#pragma unroll
  for (int it = 0; it < 4; ++it) {
    const int idx = tid + it * 256;
    const int r = idx >> 5, c = idx & 31;
    O[(size_t)(d0 + r) * DIM + o0 + c] = t[c * 33 + r];  // write along o
  }
}

// ---------------------------------------------------------------------------
// Kernel 1: QKV projection.  Out[m, o] = sum_d xh[m, d] * Wt[d, o]
//   grid = (N/128, M/128, 3), block = 256 (8 waves), tile 128x128, BK = 32.
//   Wave grid 2(M) x 4(N): each wave 64x32 -> 8 WMMAs / k-step, acc 64 VGPRs.
//   A fragments are ds_load_b128, B fragments strided gathers; tiles staged async.
// ---------------------------------------------------------------------------

#define PLDA 40   // sXh row pitch (halves): 80B rows, 16B aligned
#define PLDB 136  // sWt row pitch (halves): 272B rows, 16B aligned

__global__ __launch_bounds__(256) void qkv_gemm_kernel(
    const _Float16* __restrict__ xh, const _Float16* __restrict__ Wt,
    _Float16* __restrict__ Qh, _Float16* __restrict__ Kh,
    _Float16* __restrict__ Vh) {
  __shared__ _Float16 sXh[128 * PLDA];  // 128 rows(m) x 32 cols(k)
  __shared__ _Float16 sWt[32 * PLDB];   // 32 rows(k) x 128 cols(n)

  const int tid  = threadIdx.x;
  const int lane = tid & 31;
  const int wave = tid >> 5;
  const int wm   = (wave >> 2) * 64;  // wave's M offset in tile (0 or 64)
  const int wn   = (wave & 3) * 32;   // wave's N offset in tile (0..96)

  const int n0 = blockIdx.x * 128;
  const int m0 = blockIdx.y * 128;
  const _Float16* Wz = Wt + (size_t)blockIdx.z * DIM * DIM;
  _Float16* Out = (blockIdx.z == 0) ? Qh : (blockIdx.z == 1) ? Kh : Vh;

  v8f acc[4][2] = {};  // [mt][nt]

  for (int k0 = 0; k0 < DIM; k0 += 32) {
    __syncthreads();
    // Stage xh tile: 128x32 halves = 512 b128 chunks; 2 per thread.
#pragma unroll
    for (int it = 0; it < 2; ++it) {
      const int c   = tid + it * 256;   // 0..511
      const int row = c >> 2;           // 4 chunks per 32-half row
      const int off = (c & 3) * 8;
      async_b128(&xh[(size_t)(m0 + row) * DIM + k0 + off],
                 &sXh[row * PLDA + off]);
    }
    // Stage Wt tile: 32x128 halves = 512 b128 chunks; 2 per thread.
#pragma unroll
    for (int it = 0; it < 2; ++it) {
      const int c   = tid + it * 256;
      const int row = c >> 4;           // 16 chunks per 128-half row
      const int off = (c & 15) * 8;
      async_b128(&Wz[(size_t)(k0 + row) * DIM + n0 + off],
                 &sWt[row * PLDB + off]);
    }
    async_wait();
    __syncthreads();

    v16h af[4];
#pragma unroll
    for (int mt = 0; mt < 4; ++mt)
      af[mt] = frag_a16(sXh + (wm + mt * 16) * PLDA, PLDA, lane);
    v16h bf[2];
#pragma unroll
    for (int nt = 0; nt < 2; ++nt)
      bf[nt] = frag_b_nf(sWt + wn + nt * 16, PLDB, lane);
#pragma unroll
    for (int mt = 0; mt < 4; ++mt)
#pragma unroll
      for (int nt = 0; nt < 2; ++nt)
        acc[mt][nt] = wmma_f16(af[mt], bf[nt], acc[mt][nt]);
  }

  const int hi = lane >> 4;
  const int nl = lane & 15;
#pragma unroll
  for (int mt = 0; mt < 4; ++mt)
#pragma unroll
    for (int nt = 0; nt < 2; ++nt)
#pragma unroll
      for (int r = 0; r < 8; ++r) {
        const int mg = m0 + wm + mt * 16 + r + 8 * hi;
        const int ng = n0 + wn + nt * 16 + nl;
        Out[(size_t)mg * DIM + ng] = (_Float16)acc[mt][nt][r];
      }
}

// ---------------------------------------------------------------------------
// Kernel 2: causal flash attention.
//   grid = (S/16, B), block = 256 (8 waves).
//   Q kept transposed in LDS (sQt[d][q], 1024x16), K/V tiles 32x1024 f16.
//   S^T = K Q^T : K-reduction (1024) split over 8 waves, ds_add_f32 combine.
//   O += P V   : N (1024) split over 8 waves, 8 accumulator C-tiles each.
// ---------------------------------------------------------------------------

#define LDQT 24    // 48B rows: 16B aligned
#define LDK  1032  // 2064B rows: 16B aligned
#define LDV  1032
#define LDP  40
#define OFF_QT 0
#define OFF_K  49152   // 1024*24*2
#define OFF_V  115200  // + 32*1032*2
#define OFF_S  181248  // + 32*1032*2
#define OFF_P  183296  // + 16*32*4
#define OFF_M  184576  // + 16*40*2
#define OFF_L  184640
#define OFF_C  184704
#define SMEM_BYTES 184768

__global__ __launch_bounds__(256) void flash_attn_kernel(
    const _Float16* __restrict__ Qh, const _Float16* __restrict__ Kh,
    const _Float16* __restrict__ Vh, float* __restrict__ Out) {
  extern __shared__ char smem[];
  _Float16* sQt = (_Float16*)(smem + OFF_QT);  // 1024 x LDQT (d-major Q^T)
  _Float16* sK  = (_Float16*)(smem + OFF_K);   // 32 x LDK
  _Float16* sV  = (_Float16*)(smem + OFF_V);   // 32 x LDV
  float*    sS  = (float*)(smem + OFF_S);      // 16 x 32 scores [q][kv]
  _Float16* sP  = (_Float16*)(smem + OFF_P);   // 16 x LDP probs (f16)
  float*    sM  = (float*)(smem + OFF_M);      // running row max
  float*    sL  = (float*)(smem + OFF_L);      // running row sum
  float*    sSc = (float*)(smem + OFF_C);      // per-row rescale factor

  const int tid  = threadIdx.x;
  const int lane = tid & 31;
  const int wave = tid >> 5;
  const int hi   = lane >> 4;
  const int nl   = lane & 15;

  const int q0 = blockIdx.x * 16;
  const int b  = blockIdx.y;
  const size_t baseQ  = ((size_t)b * SEQ + q0) * DIM;
  const size_t baseKV = (size_t)b * SEQ * DIM;
  const int nbase = wave * 128;  // this wave's output-column base

  // Build transposed Q tile once: sQt[d][q] = Q[q][d].
#pragma unroll
  for (int it = 0; it < 8; ++it) {
    const int c  = tid + it * 256;  // 0..2047 chunks of 8 halves
    const int q  = c >> 7;
    const int d0 = (c & 127) * 8;
    const uint4 raw = *(const uint4*)&Qh[baseQ + (size_t)q * DIM + d0];
    const _Float16* h = (const _Float16*)&raw;
#pragma unroll
    for (int e = 0; e < 8; ++e) sQt[(d0 + e) * LDQT + q] = h[e];
  }
  if (tid < 16) { sM[tid] = -INFINITY; sL[tid] = 0.0f; }

  v8f acc[8] = {};  // 16 x 128 fp32 output accumulator per wave

  const int ntiles = (q0 + 15) / 32 + 1;  // causal: only k <= q0+15 needed
  for (int j = 0; j < ntiles; ++j) {
    const int k0 = j * 32;
    __syncthreads();  // previous iteration done with sK/sV/sP
    sS[tid] = 0.0f;
    sS[tid + 256] = 0.0f;
    // Stream K and V tiles (32 x 1024 f16 each) into LDS (async).
#pragma unroll
    for (int it = 0; it < 16; ++it) {
      const int c   = tid + it * 256;
      const int row = c >> 7;
      const int off = (c & 127) * 8;
      const size_t g = baseKV + (size_t)(k0 + row) * DIM + off;
      async_b128(&Kh[g], &sK[row * LDK + off]);
      async_b128(&Vh[g], &sV[row * LDV + off]);
    }
    async_wait();
    __syncthreads();

    // ---- S^T = K Q^T : wave reduces K-dim chunk [wave*128, +128) ----
    v8f t0 = {}, t1 = {};
#pragma unroll
    for (int kk = 0; kk < 4; ++kk) {
      const int kc = wave * 128 + kk * 32;
      const v16h ak0 = frag_a16(sK + kc, LDK, lane);             // kv rows 0..15
      const v16h ak1 = frag_a16(sK + 16 * LDK + kc, LDK, lane);  // kv rows 16..31
      const v16h bq  = frag_b_nf(sQt + (size_t)kc * LDQT, LDQT, lane);
      t0 = wmma_f16(ak0, bq, t0);
      t1 = wmma_f16(ak1, bq, t1);
    }
    // C-tile of S^T: M = kv = r + 8*hi (+16 for t1), N = q = nl.
#pragma unroll
    for (int r = 0; r < 8; ++r) {
      atomicAdd(&sS[nl * 32 + r + 8 * hi], t0[r]);
      atomicAdd(&sS[nl * 32 + 16 + r + 8 * hi], t1[r]);
    }
    __syncthreads();

    // ---- online softmax update, one thread per q row ----
    if (tid < 16) {
      const float scale = 0.03125f;  // 1/sqrt(1024)
      const int r = tid;
      const int qidx = q0 + r;
      const float mOld = sM[r];
      float mNew = mOld;
      for (int c = 0; c < 32; ++c)
        if (k0 + c <= qidx) mNew = fmaxf(mNew, sS[r * 32 + c] * scale);
      const float corr =
          (mOld == -INFINITY) ? ((mNew == -INFINITY) ? 1.0f : 0.0f)
                              : __expf(mOld - mNew);
      float lsum = 0.0f;
      for (int c = 0; c < 32; ++c) {
        float p = 0.0f;
        if (k0 + c <= qidx) p = __expf(sS[r * 32 + c] * scale - mNew);
        sP[r * LDP + c] = (_Float16)p;
        lsum += p;
      }
      sM[r]  = mNew;
      sL[r]  = sL[r] * corr + lsum;
      sSc[r] = corr;
    }
    __syncthreads();

    // ---- rescale accumulators, then O += P * V ----
    float scr[8];
#pragma unroll
    for (int r = 0; r < 8; ++r) scr[r] = sSc[r + 8 * hi];
#pragma unroll
    for (int t = 0; t < 8; ++t)
#pragma unroll
      for (int r = 0; r < 8; ++r) acc[t][r] *= scr[r];

    const v16h ap = frag_a16(sP, LDP, lane);
#pragma unroll
    for (int t = 0; t < 8; ++t) {
      const v16h bv = frag_b_nf(sV + nbase + t * 16, LDV, lane);
      acc[t] = wmma_f16(ap, bv, acc[t]);
    }
  }

  // ---- epilogue: divide by l, store fp32 ----
  float linv[8];
#pragma unroll
  for (int r = 0; r < 8; ++r) linv[r] = 1.0f / sL[r + 8 * hi];
#pragma unroll
  for (int t = 0; t < 8; ++t)
#pragma unroll
    for (int r = 0; r < 8; ++r) {
      const size_t o =
          ((size_t)b * SEQ + q0 + r + 8 * hi) * DIM + nbase + t * 16 + nl;
      Out[o] = acc[t][r] * linv[r];
    }
}

// ---------------------------------------------------------------------------

extern "C" void kernel_launch(void* const* d_in, const int* in_sizes, int n_in,
                              void* d_out, int out_size, void* d_ws,
                              size_t ws_size, hipStream_t stream) {
  const float* x  = (const float*)d_in[0];
  const float* Wq = (const float*)d_in[1];
  const float* Wk = (const float*)d_in[2];
  const float* Wv = (const float*)d_in[3];
  float* out = (float*)d_out;

  const size_t MD = (size_t)BATCH * SEQ * DIM;  // 16M elements
  const size_t WD = (size_t)DIM * DIM;          // 1M elements
  _Float16* xh = (_Float16*)d_ws;
  _Float16* Wt = xh + MD;        // 3 transposed weight matrices
  _Float16* Qh = Wt + 3 * WD;
  _Float16* Kh = Qh + MD;
  _Float16* Vh = Kh + MD;

  // Kernel 0a: x -> f16 (coalesced).
  convert_x_kernel<<<dim3(MD / 1024), dim3(256), 0, stream>>>(x, xh);
  // Kernel 0b: W -> f16 transposed [d][o].
  transpose_w_kernel<<<dim3(DIM / 32, DIM / 32, 3), dim3(256), 0, stream>>>(
      Wq, Wk, Wv, Wt);

  // Kernel 1: QKV projections (z selects which weight matrix).
  dim3 gridG(DIM / 128, (BATCH * SEQ) / 128, 3);
  qkv_gemm_kernel<<<gridG, dim3(256), 0, stream>>>(xh, Wt, Qh, Kh, Vh);

  // Kernel 2: flash attention, ~185KB dynamic LDS (CDNA5 has 320KB/WGP).
  (void)hipFuncSetAttribute((const void*)flash_attn_kernel,
                            hipFuncAttributeMaxDynamicSharedMemorySize,
                            SMEM_BYTES);
  dim3 gridA(SEQ / 16, BATCH);
  flash_attn_kernel<<<gridA, dim3(256), SMEM_BYTES, stream>>>(Qh, Kh, Vh, out);
}
